// NODE_17729624998413
// MI455X (gfx1250) — matrix operand order repaired and checked by reading, not compiled, not run on hardware
//
#include <hip/hip_runtime.h>
#include <math.h>

// ---------------------------------------------------------------------------
// Types for CDNA5 WMMA (wave32). bf16 A/B, f32 accumulate, 16x16x32 shape.
// ---------------------------------------------------------------------------
typedef __attribute__((ext_vector_type(16))) __bf16  v16bf;
typedef __attribute__((ext_vector_type(8)))  float   v8f;
typedef __attribute__((ext_vector_type(4)))  unsigned u32x4;
typedef __attribute__((ext_vector_type(8)))  int      i32x8;
typedef __attribute__((ext_vector_type(4)))  int      i32x4;

#if defined(__has_builtin)
#if __has_builtin(__builtin_amdgcn_tensor_load_to_lds)
#define USE_TDM 1
#endif
#endif

#define B_   512
#define S_   128
#define D_   64
#define H_   256
#define T_   16
#define INP  160      // IN=129 zero-padded to multiple of 32
#define G4H  1024     // 4*H
#define KMAX 256      // max GEMM K (W_hh)

struct GemmArgs {
  const __bf16* A;     // [M, lda] bf16
  const __bf16* Bm;    // [K, N]  bf16 (pre-transposed weights)
  const float*  bias;  // [N] or null
  const float*  Cadd;  // [M, N] addend or null
  float*        Cf;    // f32 out or null
  __bf16*       Cb;    // bf16 out or null
  int lda, N, K, relu;
};

// ---------------------------------------------------------------------------
// WMMA GEMM: grid (M/128, N/64, ndir). 256 thr = 8 waves; wave -> 16x64 strip.
// B panel [K x 64] is DMA'd once into LDS by the Tensor Data Mover (TDM),
// then restaged per 32-K chunk into a transposed, padded tile for b128 reads.
// ---------------------------------------------------------------------------
__global__ __launch_bounds__(256) void gemm_wmma(GemmArgs a0, GemmArgs a1) {
  GemmArgs ga = (blockIdx.z == 0) ? a0 : a1;
  __shared__ __align__(16) __bf16 ldsRaw[KMAX * 64];  // row-major B panel (TDM dest)
  __shared__ __align__(16) __bf16 ldsB[64 * 40];      // B chunk transposed [64][32], pitch 40

  const int tid  = threadIdx.x;
  const int w    = tid >> 5;
  const int lane = tid & 31;
  const int wgM  = blockIdx.x * 128;
  const int n0   = blockIdx.y * 64;

  const int rowA = wgM + w * 16 + (lane & 15);
  const int kh   = lane >> 4;                 // half-wave: selects K sub-blocks
  const int br   = tid >> 3;                  // stage: row 0..31
  const int bc   = (tid & 7) << 3;            // stage: col block

#ifdef USE_TDM
  if (tid < 32) {
    // Tensor DMA descriptor (D#), CDNA5 ISA: 2-D bf16 tile,
    // tile_dim0 = 64 cols (contiguous), tile_dim1 = K rows, row stride = N.
    unsigned long long ga64 = (unsigned long long)(uintptr_t)(ga.Bm + n0);
    unsigned ldsoff = (unsigned)(uintptr_t)(&ldsRaw[0]);   // LDS byte offset = addr[31:0]
    u32x4 g0;
    g0[0] = 1u;                                        // count=1, user mode
    g0[1] = ldsoff;                                    // lds_addr
    g0[2] = (unsigned)(ga64 & 0xffffffffu);            // global_addr[31:0]
    g0[3] = (unsigned)((ga64 >> 32) & 0x01ffffffu) | (2u << 30);  // addr[56:32] | type=2
    unsigned dim0 = (unsigned)(ga.N - n0);             // >= 64, tile fully in-bounds
    unsigned Ku   = (unsigned)ga.K;
    unsigned long long strd = (unsigned long long)ga.N;
    i32x8 g1;
    g1[0] = (int)(1u << 16);                                   // data_size = 2 bytes
    g1[1] = (int)((dim0 & 0xffffu) << 16);                     // tensor_dim0[15:0]
    g1[2] = (int)((dim0 >> 16) | ((Ku & 0xffffu) << 16));      // dim0[31:16] | dim1[15:0]
    g1[3] = (int)((Ku >> 16) | (64u << 16));                   // dim1[31:16] | tile_dim0=64
    g1[4] = (int)(Ku & 0xffffu);                               // tile_dim1=K, tile_dim2=0
    g1[5] = (int)(unsigned)(strd & 0xffffffffu);               // tensor_dim0_stride[31:0]
    g1[6] = (int)(unsigned)((strd >> 32) & 0xffffu);           // stride[47:32] | d1stride=0
    g1[7] = 0;
    i32x4 gz = {0, 0, 0, 0};
#if __clang_major__ >= 23
    i32x8 gz8 = {0, 0, 0, 0, 0, 0, 0, 0};
    __builtin_amdgcn_tensor_load_to_lds(g0, g1, gz, gz, gz8, 0);
#else
    __builtin_amdgcn_tensor_load_to_lds(g0, g1, gz, gz, 0);
#endif
    __builtin_amdgcn_s_wait_tensorcnt(0);
  }
  __syncthreads();
#endif

  v8f acc[4] = {};

  for (int kc = 0; kc < ga.K; kc += 32) {
    // Restage B[kc..kc+31][n0..n0+63] transposed into ldsB.
    union { u32x4 u; __bf16 h[8]; } ld;
#ifdef USE_TDM
    ld.u = *(const u32x4*)(&ldsRaw[(kc + br) * 64 + bc]);
#else
    ld.u = *(const u32x4*)(ga.Bm + (size_t)(kc + br) * ga.N + n0 + bc);
#endif
#pragma unroll
    for (int i = 0; i < 8; ++i) ldsB[(bc + i) * 40 + br] = ld.h[i];
    __syncthreads();

    // A fragment per ISA 16-bit A 16x32 layout (lane<16: K 0-7,16-23; else 8-15,24-31).
    union { u32x4 u[2]; v16bf v; } fa;
    const __bf16* Ap = ga.A + (size_t)rowA * ga.lda + kc + (kh << 3);
    fa.u[0] = *(const u32x4*)(Ap);
    fa.u[1] = *(const u32x4*)(Ap + 16);
    if (kc + 32 < ga.K)
      __builtin_prefetch(ga.A + (size_t)rowA * ga.lda + kc + 32, 0, 1);

#pragma unroll
    for (int j = 0; j < 4; ++j) {
      union { u32x4 u[2]; v16bf v; } fb;
      const __bf16* bp = &ldsB[((lane & 15) + j * 16) * 40 + (kh << 3)];
      fb.u[0] = *(const u32x4*)(bp);
      fb.u[1] = *(const u32x4*)(bp + 16);
      acc[j] = __builtin_amdgcn_wmma_f32_16x16x32_bf16(
          false, fa.v, false, fb.v, (short)0, acc[j], false, false);
    }
    __syncthreads();
  }

  // D layout: VGPR r -> M = r (lanes<16) or r+8 (lanes>=16); N = lane&15.
  const int colb = n0 + (lane & 15);
  const int rowb = wgM + w * 16 + (kh << 3);
#pragma unroll
  for (int j = 0; j < 4; ++j) {
#pragma unroll
    for (int r = 0; r < 8; ++r) {
      const int row = rowb + r;
      const int col = colb + j * 16;
      const size_t o = (size_t)row * ga.N + col;
      float v = acc[j][r];
      if (ga.bias) v += ga.bias[col];
      if (ga.Cadd) v += ga.Cadd[o];
      if (ga.relu) v = fmaxf(v, 0.f);
      if (ga.Cf) ga.Cf[o] = v;
      if (ga.Cb) ga.Cb[o] = (__bf16)v;
    }
  }
}

// ---------------------------------------------------------------------------
// Converters
// ---------------------------------------------------------------------------
__global__ void zero_u32(unsigned* p, int n) {
  int i = blockIdx.x * blockDim.x + threadIdx.x;
  if (i < n) p[i] = 0u;
}

// W [N, Ksrc] row-major -> out [Kout, N] bf16, zero-padded K rows.
__global__ void transpose_bf16(const float* in, __bf16* out, int N, int Kout, int Ksrc) {
  int idx = blockIdx.x * blockDim.x + threadIdx.x;
  if (idx >= N * Kout) return;
  int k = idx / N, n = idx - k * N;
  out[idx] = (k < Ksrc) ? (__bf16)in[(size_t)n * Ksrc + k] : (__bf16)0.f;
}

// x [B,S,129] f32 -> xb [S*B, 160] bf16 (row m = s*512 + b), K padded with 0.
__global__ void convert_xb(const float* x, __bf16* xb) {
  int idx = blockIdx.x * blockDim.x + threadIdx.x;
  if (idx >= S_ * B_ * INP) return;
  int m = idx / INP, k = idx - m * INP;
  int s = m >> 9, b = m & (B_ - 1);
  xb[idx] = (k < 129) ? (__bf16)x[((size_t)b * S_ + s) * 129 + k] : (__bf16)0.f;
}

// ---------------------------------------------------------------------------
// LSTM gate fusion for both directions. g = [B,4H] pre-activation.
// ---------------------------------------------------------------------------
__global__ void lstm_gates(const float* g_f, const float* g_b,
                           float* h_f, float* c_f, float* h_b, float* c_b,
                           __bf16* hb_f, __bf16* hb_b,
                           float* fwd, const int* lengths, int s) {
  int idx = blockIdx.x * blockDim.x + threadIdx.x;   // 0 .. 2*B*H-1
  int dir = idx >> 17;
  int r   = idx & (B_ * H_ - 1);
  int b   = r >> 8;
  int j   = r & (H_ - 1);
  const float* g = dir ? g_b : g_f;
  float* h = dir ? h_b : h_f;
  float* c = dir ? c_b : c_f;
  __bf16* hb = dir ? hb_b : hb_f;

  const size_t base = (size_t)b * G4H;
  float gi = g[base + j];
  float gf = g[base + H_ + j];
  float gg = g[base + 2 * H_ + j];
  float go = g[base + 3 * H_ + j];
  float si = 1.f / (1.f + expf(-gi));
  float sf = 1.f / (1.f + expf(-gf));
  float so = 1.f / (1.f + expf(-go));
  float cn = sf * c[r] + si * tanhf(gg);
  float hn = so * tanhf(cn);
  c[r] = cn;
  h[r] = hn;
  hb[r] = (__bf16)hn;
  if (!dir && s == lengths[b] - 1) fwd[r] = hn;
}

__global__ void combine_hidden(const float* fwd, const float* bwd, __bf16* hiddenb) {
  int i = blockIdx.x * blockDim.x + threadIdx.x;
  hiddenb[i] = (__bf16)(0.5f * (fwd[i] + bwd[i]));
}

// ---------------------------------------------------------------------------
// Neural-ODE RK4: one workgroup per batch element, weights in registers.
// deriv(z) = relu(z @ Wd1^T + bd1) @ Wd2^T + bd2 ; 15 intervals x 8 substeps.
// ---------------------------------------------------------------------------
__global__ __launch_bounds__(256) void ode_kernel(const float* y0, const float* lt,
                                                  const float* Wd1, const float* bd1,
                                                  const float* Wd2, const float* bd2,
                                                  float* predict) {
  const int b = blockIdx.x, tid = threadIdx.x;
  __shared__ float ys[64], zs[64], hs[256], part[256], kk[4][64];

  float w1[64], w2[64];
#pragma unroll
  for (int k = 0; k < 64; ++k) w1[k] = Wd1[(size_t)tid * 64 + k];
  const int oi = tid & 63, oq = tid >> 6;
#pragma unroll
  for (int j = 0; j < 64; ++j) w2[j] = Wd2[(size_t)oi * 256 + oq * 64 + j];
  const float b1 = bd1[tid];
  const float b2 = bd2[oi];

  if (tid < 64) {
    float v = y0[(size_t)b * 64 + tid];
    ys[tid] = v;
    predict[(size_t)b * T_ * 64 + tid] = v;
  }
  __syncthreads();

  auto deriv = [&](const float* zz, float* outp) {
    float acc = b1;
#pragma unroll
    for (int k = 0; k < 64; ++k) acc += zz[k] * w1[k];
    hs[tid] = fmaxf(acc, 0.f);
    __syncthreads();
    float p = 0.f;
#pragma unroll
    for (int j = 0; j < 64; ++j) p += hs[oq * 64 + j] * w2[j];
    part[tid] = p;
    __syncthreads();
    if (tid < 64) outp[tid] = part[tid] + part[64 + tid] + part[128 + tid] + part[192 + tid] + b2;
    __syncthreads();
  };

  for (int t = 0; t < T_ - 1; ++t) {
    const float dt = (lt[b * T_ + t + 1] - lt[b * T_ + t]) * 0.125f;  // /K, K=8
    for (int ss = 0; ss < 8; ++ss) {
      deriv(ys, kk[0]);
      if (tid < 64) zs[tid] = ys[tid] + 0.5f * dt * kk[0][tid];
      __syncthreads();
      deriv(zs, kk[1]);
      if (tid < 64) zs[tid] = ys[tid] + 0.5f * dt * kk[1][tid];
      __syncthreads();
      deriv(zs, kk[2]);
      if (tid < 64) zs[tid] = ys[tid] + dt * kk[2][tid];
      __syncthreads();
      deriv(zs, kk[3]);
      if (tid < 64)
        ys[tid] += (dt * (1.f / 6.f)) *
                   (kk[0][tid] + 2.f * kk[1][tid] + 2.f * kk[2][tid] + kk[3][tid]);
      __syncthreads();
    }
    if (tid < 64) predict[((size_t)b * T_ + t + 1) * 64 + tid] = ys[tid];
    __syncthreads();
  }
}

// ---------------------------------------------------------------------------
// Deterministic masked-MSE reduction (no float atomics -> replay-stable).
// ---------------------------------------------------------------------------
__global__ void loss_reduce(const float* pred, const float* lf, const float* mask,
                            float* partials) {
  __shared__ float sd[256];
  int i = blockIdx.x * 256 + threadIdx.x;
  float d = pred[i] - lf[i];
  sd[threadIdx.x] = d * d * (1.f - mask[i]);
  __syncthreads();
  for (int s = 128; s > 0; s >>= 1) {
    if (threadIdx.x < s) sd[threadIdx.x] += sd[threadIdx.x + s];
    __syncthreads();
  }
  if (threadIdx.x == 0) partials[blockIdx.x] = sd[0];
}

__global__ void loss_final(const float* partials, float* out) {
  __shared__ float sd[256];
  int tid = threadIdx.x;
  float a = 0.f;
  for (int i = tid; i < 2048; i += 256) a += partials[i];
  sd[tid] = a;
  __syncthreads();
  for (int s = 128; s > 0; s >>= 1) {
    if (tid < s) sd[tid] += sd[tid + s];
    __syncthreads();
  }
  if (tid == 0) out[0] = sd[0] * (1.f / 524288.f);
}

// ---------------------------------------------------------------------------
// Host driver
// ---------------------------------------------------------------------------
extern "C" void kernel_launch(void* const* d_in, const int* in_sizes, int n_in,
                              void* d_out, int out_size, void* d_ws, size_t ws_size,
                              hipStream_t stream) {
  (void)in_sizes; (void)n_in; (void)out_size; (void)ws_size;

  const float* x     = (const float*)d_in[0];
  const int*   lens  = (const int*)d_in[1];
  const float* lf    = (const float*)d_in[2];
  const float* lt    = (const float*)d_in[3];
  const float* mask  = (const float*)d_in[4];
  const float* Wih_f = (const float*)d_in[5];
  const float* Whh_f = (const float*)d_in[6];
  const float* b_f   = (const float*)d_in[7];
  const float* Wih_b = (const float*)d_in[8];
  const float* Whh_b = (const float*)d_in[9];
  const float* b_b   = (const float*)d_in[10];
  const float* Wp1   = (const float*)d_in[11];
  const float* bp1   = (const float*)d_in[12];
  const float* Wp2   = (const float*)d_in[13];
  const float* bp2   = (const float*)d_in[14];
  const float* Wd1   = (const float*)d_in[15];
  const float* bd1   = (const float*)d_in[16];
  const float* Wd2   = (const float*)d_in[17];
  const float* bd2   = (const float*)d_in[18];

  // ---- workspace carve (~30 MB total; fully L2-resident on MI455X) ----
  char* ws = (char*)d_ws;
  size_t off = 0;
  auto carve = [&](size_t bytes) -> char* {
    char* p = ws + off;
    off += (bytes + 255) & ~(size_t)255;
    return p;
  };
  __bf16* xb     = (__bf16*)carve((size_t)S_ * B_ * INP * 2);   // 21.0 MB
  __bf16* WihT_f = (__bf16*)carve((size_t)INP * G4H * 2);
  __bf16* WihT_b = (__bf16*)carve((size_t)INP * G4H * 2);
  __bf16* WhhT_f = (__bf16*)carve((size_t)H_ * G4H * 2);
  __bf16* WhhT_b = (__bf16*)carve((size_t)H_ * G4H * 2);
  __bf16* Wp1T   = (__bf16*)carve((size_t)H_ * H_ * 2);
  __bf16* Wp2T   = (__bf16*)carve((size_t)H_ * D_ * 2);
  float*  g_f    = (float*)carve((size_t)B_ * G4H * 4);
  float*  g_b    = (float*)carve((size_t)B_ * G4H * 4);
  char*   zbase  = ws + off;                                    // zero-init region:
  float*  h_f    = (float*)carve((size_t)B_ * H_ * 4);
  float*  c_f    = (float*)carve((size_t)B_ * H_ * 4);
  float*  h_b    = (float*)carve((size_t)B_ * H_ * 4);
  float*  c_b    = (float*)carve((size_t)B_ * H_ * 4);
  __bf16* hb_f   = (__bf16*)carve((size_t)B_ * H_ * 2);
  __bf16* hb_b   = (__bf16*)carve((size_t)B_ * H_ * 2);
  int zdw = (int)(((ws + off) - zbase) / 4);
  float*  fwd    = (float*)carve((size_t)B_ * H_ * 4);
  __bf16* hidb   = (__bf16*)carve((size_t)B_ * H_ * 2);
  __bf16* t1b    = (__bf16*)carve((size_t)B_ * H_ * 2);
  float*  y0     = (float*)carve((size_t)B_ * D_ * 4);
  float*  lpart  = (float*)carve(2048 * 4);

  float* pred    = (float*)d_out;                 // [B,T,D]
  float* lossout = pred + (size_t)B_ * T_ * D_;   // scalar

  // ---- prep: zero state, convert inputs/weights to bf16 layouts ----
  zero_u32<<<(zdw + 255) / 256, 256, 0, stream>>>((unsigned*)zbase, zdw);
  convert_xb<<<(S_ * B_ * INP + 255) / 256, 256, 0, stream>>>(x, xb);
  transpose_bf16<<<(G4H * INP + 255) / 256, 256, 0, stream>>>(Wih_f, WihT_f, G4H, INP, 129);
  transpose_bf16<<<(G4H * INP + 255) / 256, 256, 0, stream>>>(Wih_b, WihT_b, G4H, INP, 129);
  transpose_bf16<<<(G4H * H_ + 255) / 256, 256, 0, stream>>>(Whh_f, WhhT_f, G4H, H_, H_);
  transpose_bf16<<<(G4H * H_ + 255) / 256, 256, 0, stream>>>(Whh_b, WhhT_b, G4H, H_, H_);
  transpose_bf16<<<(H_ * H_ + 255) / 256, 256, 0, stream>>>(Wp1, Wp1T, H_, H_, H_);
  transpose_bf16<<<(D_ * H_ + 255) / 256, 256, 0, stream>>>(Wp2, Wp2T, D_, H_, H_);

  // ---- bidirectional LSTM scan (fused input-proj + recurrent GEMM + gates) ----
  for (int s = 0; s < S_; ++s) {
    GemmArgs ax_f; ax_f.A = xb + (size_t)s * B_ * INP;        ax_f.Bm = WihT_f;
    ax_f.bias = b_f; ax_f.Cadd = nullptr; ax_f.Cf = g_f; ax_f.Cb = nullptr;
    ax_f.lda = INP; ax_f.N = G4H; ax_f.K = INP; ax_f.relu = 0;
    GemmArgs ax_b = ax_f; ax_b.A = xb + (size_t)(S_ - 1 - s) * B_ * INP;
    ax_b.Bm = WihT_b; ax_b.bias = b_b; ax_b.Cf = g_b;
    gemm_wmma<<<dim3(4, 16, 2), 256, 0, stream>>>(ax_f, ax_b);

    GemmArgs ah_f; ah_f.A = hb_f; ah_f.Bm = WhhT_f;
    ah_f.bias = nullptr; ah_f.Cadd = g_f; ah_f.Cf = g_f; ah_f.Cb = nullptr;
    ah_f.lda = H_; ah_f.N = G4H; ah_f.K = H_; ah_f.relu = 0;
    GemmArgs ah_b = ah_f; ah_b.A = hb_b; ah_b.Bm = WhhT_b; ah_b.Cadd = g_b; ah_b.Cf = g_b;
    gemm_wmma<<<dim3(4, 16, 2), 256, 0, stream>>>(ah_f, ah_b);

    lstm_gates<<<(2 * B_ * H_) / 256, 256, 0, stream>>>(
        g_f, g_b, h_f, c_f, h_b, c_b, hb_f, hb_b, fwd, lens, s);
  }

  // ---- head: hidden = 0.5*(fwd+bwd); y0 = relu(hidden@Wp1^T+bp1)@Wp2^T+bp2 ----
  combine_hidden<<<(B_ * H_) / 256, 256, 0, stream>>>(fwd, h_b, hidb);

  GemmArgs p1; p1.A = hidb; p1.Bm = Wp1T; p1.bias = bp1; p1.Cadd = nullptr;
  p1.Cf = nullptr; p1.Cb = t1b; p1.lda = H_; p1.N = H_; p1.K = H_; p1.relu = 1;
  gemm_wmma<<<dim3(4, 4, 1), 256, 0, stream>>>(p1, p1);

  GemmArgs p2; p2.A = t1b; p2.Bm = Wp2T; p2.bias = bp2; p2.Cadd = nullptr;
  p2.Cf = y0; p2.Cb = nullptr; p2.lda = H_; p2.N = D_; p2.K = H_; p2.relu = 0;
  gemm_wmma<<<dim3(4, 1, 1), 256, 0, stream>>>(p2, p2);

  // ---- neural ODE RK4 + loss ----
  ode_kernel<<<B_, 256, 0, stream>>>(y0, lt, Wd1, bd1, Wd2, bd2, pred);
  loss_reduce<<<2048, 256, 0, stream>>>(pred, lf, mask, lpart);
  loss_final<<<1, 256, 0, stream>>>(lpart, lossout);
}
